// EGNN_35150012351089
// MI455X (gfx1250) — matrix operand order repaired and checked by reading, not compiled, not run on hardware
//
#include <hip/hip_runtime.h>

// ---------------------------------------------------------------------------
// EGNN forward for MI455X (gfx1250, wave32, WMMA bf16).
//
// Heavy GEMMs run through v_wmma_f32_16x16x32_bf16 with fp32 accumulation.
// A-matrix rows are gathered on the fly (h[row] | h[col] | edge_attr) into
// LDS; weights are pre-converted to bf16 in [dout][din] layout, which is the
// [N][K] layout the B-stage consumes directly (no transpose needed).
//
// Assumed d_in layout (77 entries):
//   0:h 1:pos 2:edge_index 3:edge_attr 4:node_mask 5:edge_mask
//   then params flattened jax-pytree style (dict keys sorted):
//   blocks[0..2]: coordW0,coordb0,coordW1,coordb1,coordW2,
//                 gcl0{eW0,eb0,eW1,eb1,nW0,nb0,nW1,nb1}, gcl1{...}   (21 each)
//   69:edge_embedding.W 70:.b  71:edge_embedding_out.W 72:.b
//   73:embedding.W 74:.b       75:embedding_out.W 76:.b
// ---------------------------------------------------------------------------

#define NN   10000
#define NE   160000
#define HIDD 256

typedef __attribute__((ext_vector_type(16))) __bf16 v16bf;
typedef __attribute__((ext_vector_type(8)))  float  v8f;

struct U128 { unsigned x, y, z, w; };
union Frag { U128 u[2]; v16bf v; };

__device__ __forceinline__ float silu_f(float x) { return x / (1.f + __expf(-x)); }

// ---------------- small utility kernels ----------------
__global__ void k_f2bf(const float* __restrict__ s, __bf16* __restrict__ d, int n) {
  int i = blockIdx.x * blockDim.x + threadIdx.x;
  if (i < n) d[i] = (__bf16)s[i];
}
__global__ void k_zero(float* __restrict__ p, int n) {
  int i = blockIdx.x * blockDim.x + threadIdx.x;
  if (i < n) p[i] = 0.f;
}
__global__ void k_copy(const float* __restrict__ s, float* __restrict__ d, int n) {
  int i = blockIdx.x * blockDim.x + threadIdx.x;
  if (i < n) d[i] = s[i];
}

// h = h_in @ Wemb^T + b ; also bf16 mirror.  W:(256,8)
__global__ void k_node_embed(const float* __restrict__ hin, const float* __restrict__ W,
                             const float* __restrict__ b, float* __restrict__ h,
                             __bf16* __restrict__ hb) {
  int i = blockIdx.x * 256 + threadIdx.x;
  if (i >= NN * HIDD) return;
  int n = i >> 8, c = i & 255;
  float v = b[c];
#pragma unroll
  for (int j = 0; j < 8; j++) v += hin[n * 8 + j] * W[c * 8 + j];
  h[i] = v; hb[i] = (__bf16)v;
}

// per-edge radial + coord_diff (norm_constant=1); writes ea[:,0] = bf16(radial)
__global__ void k_radial(const float* __restrict__ pos, const int* __restrict__ ei,
                         float* __restrict__ radial, float* __restrict__ cdiff,
                         __bf16* __restrict__ ea) {
  int e = blockIdx.x * blockDim.x + threadIdx.x;
  if (e >= NE) return;
  int r = ei[e], c = ei[NE + e];
  float d0 = pos[r * 3 + 0] - pos[c * 3 + 0];
  float d1 = pos[r * 3 + 1] - pos[c * 3 + 1];
  float d2 = pos[r * 3 + 2] - pos[c * 3 + 2];
  float r2 = d0 * d0 + d1 * d1 + d2 * d2;
  float inv = 1.f / (sqrtf(r2 + 1e-8f) + 1.f);
  radial[e] = r2;
  cdiff[e * 3 + 0] = d0 * inv;
  cdiff[e * 3 + 1] = d1 * inv;
  cdiff[e * 3 + 2] = d2 * inv;
  ea[(size_t)e * HIDD] = (__bf16)r2;
}

// edge_embedding: ea[:,1:256] = [radial, ea_in] @ W^T + b.  W:(255,3)
__global__ void k_edge_embed(const float* __restrict__ radial, const float* __restrict__ ein,
                             const float* __restrict__ W, const float* __restrict__ b,
                             __bf16* __restrict__ ea) {
  int i = blockIdx.x * 256 + threadIdx.x;
  int e = i >> 8, c = i & 255;
  if (e >= NE || c == 0) return;
  int j = c - 1;
  float v = W[j * 3 + 0] * radial[e] + W[j * 3 + 1] * ein[e * 2 + 0] +
            W[j * 3 + 2] * ein[e * 2 + 1] + b[j];
  ea[(size_t)e * HIDD + c] = (__bf16)v;
}

// agg[row[e]] += mij[e]*edge_mask[e]/100
__global__ void k_seg(const __bf16* __restrict__ mij, const int* __restrict__ erow,
                      const float* __restrict__ emask, float* __restrict__ agg) {
  int i = blockIdx.x * 256 + threadIdx.x;
  int e = i >> 8, c = i & 255;
  if (e >= NE) return;
  float v = (float)mij[(size_t)i] * emask[e] * 0.01f;
  atomicAdd(&agg[(size_t)erow[e] * HIDD + c], v);
}

// h = (h + upd) * node_mask ; refresh bf16 mirror
__global__ void k_node_update(const float* __restrict__ upd, const float* __restrict__ nmask,
                              float* __restrict__ h, __bf16* __restrict__ hb) {
  int i = blockIdx.x * 256 + threadIdx.x;
  if (i >= NN * HIDD) return;
  float v = (h[i] + upd[i]) * nmask[i >> 8];
  h[i] = v; hb[i] = (__bf16)v;
}

// coord update: one wave per edge. phi = t2[e] . w2 ; pos[row] += cdiff*phi*mask/100
__global__ void k_coord(const __bf16* __restrict__ t2, const float* __restrict__ w2,
                        const int* __restrict__ ei, const float* __restrict__ cdiff,
                        const float* __restrict__ emask, float* __restrict__ pos) {
  int wave = threadIdx.x >> 5, lane = threadIdx.x & 31;
  int e = blockIdx.x * 8 + wave;
  if (e >= NE) return;
  U128 u = *(const U128*)(t2 + (size_t)e * HIDD + lane * 8);
  const __bf16* bp = (const __bf16*)&u;
  float s = 0.f;
#pragma unroll
  for (int i = 0; i < 8; i++) s += (float)bp[i] * w2[lane * 8 + i];
#pragma unroll
  for (int off = 16; off > 0; off >>= 1) s += __shfl_xor(s, off, 32);
  if (lane == 0) {
    float f = s * emask[e] * 0.01f;
    int r = ei[e];
    atomicAdd(&pos[r * 3 + 0], cdiff[e * 3 + 0] * f);
    atomicAdd(&pos[r * 3 + 1], cdiff[e * 3 + 1] * f);
    atomicAdd(&pos[r * 3 + 2], cdiff[e * 3 + 2] * f);
  }
}

// out_h = (h @ W^T + b) * node_mask.  W:(8,256)
__global__ void k_out_h(const float* __restrict__ h, const float* __restrict__ W,
                        const float* __restrict__ b, const float* __restrict__ nmask,
                        float* __restrict__ out) {
  int i = blockIdx.x * blockDim.x + threadIdx.x;
  if (i >= NN * 8) return;
  int n = i >> 3, c = i & 7;
  float v = b[c];
  for (int k = 0; k < HIDD; k++) v += h[n * HIDD + k] * W[c * HIDD + k];
  out[i] = v * nmask[n];
}

// out_e = (ea[:,1:] @ W^T + b) * edge_mask.  W:(3,255)
__global__ void k_out_edge(const __bf16* __restrict__ ea, const float* __restrict__ W,
                           const float* __restrict__ b, const float* __restrict__ emask,
                           float* __restrict__ out) {
  int i = blockIdx.x * blockDim.x + threadIdx.x;
  if (i >= NE * 3) return;
  int e = i / 3, c = i % 3;
  float v = b[c];
  for (int j = 0; j < 255; j++)
    v += (float)ea[(size_t)e * HIDD + 1 + j] * W[c * 255 + j];
  out[i] = v * emask[e];
}

// ---------------- WMMA GEMM ----------------
// C[M,256] = act( gather_A[M,K] @ Wt[K,256] + bias )
// MODE 0: A = [h[row] | h[col] | ea]  (K=768)
// MODE 1: A = [h | agg]               (K=512)
// MODE 2: A = s0                      (K=256)
// Block: 256 threads = 8 waves (2x4), wave tile 32x64, k-step 32.
template <int MODE, int ACT, int OUTBF>
__global__ __launch_bounds__(256) void k_gemm(
    int M, const __bf16* __restrict__ s0, const __bf16* __restrict__ s1,
    const __bf16* __restrict__ s2, const int* __restrict__ erow,
    const int* __restrict__ ecol, const __bf16* __restrict__ Wt,
    const float* __restrict__ bias, __bf16* __restrict__ outb,
    float* __restrict__ outf) {
  constexpr int K = (MODE == 0) ? 768 : (MODE == 1) ? 512 : 256;
  constexpr int AP = 40, BP = 40;  // padded LDS strides (break bank conflicts)
  __shared__ __align__(16) __bf16 Alds[64 * AP];
  __shared__ __align__(16) __bf16 Blds[256 * BP];

  const int tid = threadIdx.x;
  const int lane = tid & 31, wave = tid >> 5;
  const int wm = wave >> 2, wn = wave & 3;  // 2 x 4 wave grid
  const int mbase = blockIdx.x * 64;

  // staging roles: 4 threads per A row, 8 bf16 (one b128) each
  const int sr = tid >> 2, sseg = tid & 3;
  const int e = mbase + sr;
  int gr = 0, gc = 0;
  if (MODE == 0 && e < M) { gr = erow[e]; gc = ecol[e]; }

  v8f acc[2][4] = {};

  for (int k0 = 0; k0 < K; k0 += 32) {
    // ---- stage A tile (64 x 32) ----
    U128 av = {0u, 0u, 0u, 0u};
    if (e < M) {
      int k = k0 + sseg * 8;
      const __bf16* p;
      if (MODE == 0) {
        p = (k < 256) ? (s0 + (size_t)gr * 256 + k)
          : (k < 512) ? (s0 + (size_t)gc * 256 + (k - 256))
                      : (s2 + (size_t)e * 256 + (k - 512));
      } else if (MODE == 1) {
        p = (k < 256) ? (s0 + (size_t)e * 256 + k)
                      : (s1 + (size_t)e * 256 + (k - 256));
      } else {
        p = s0 + (size_t)e * 256 + k;
      }
      av = *(const U128*)p;
    }
    *(U128*)&Alds[sr * AP + sseg * 8] = av;

    // ---- stage B tile (256 n-rows x 32 k) : Wt is [256][K] row-major ----
    {
      const __bf16* bp = Wt + (size_t)tid * K + k0;
      U128 b0 = *(const U128*)(bp + 0);
      U128 b1 = *(const U128*)(bp + 8);
      U128 b2 = *(const U128*)(bp + 16);
      U128 b3 = *(const U128*)(bp + 24);
      U128* q = (U128*)&Blds[tid * BP];
      q[0] = b0; q[1] = b1; q[2] = b2; q[3] = b3;
    }
    __syncthreads();

    // ---- fragments (ISA 7.12.2 layouts) ----
    const int hh = lane >> 4, l = lane & 15;
    Frag a[2], bfr[4];
#pragma unroll
    for (int tm = 0; tm < 2; tm++) {
      int m = wm * 32 + tm * 16 + l;
      a[tm].u[0] = *(const U128*)&Alds[m * AP + 8 * hh];        // K = 8h..8h+7
      a[tm].u[1] = *(const U128*)&Alds[m * AP + 16 + 8 * hh];   // K = 16+8h..
    }
#pragma unroll
    for (int tn = 0; tn < 4; tn++) {
      int n = wn * 64 + tn * 16 + l;
      bfr[tn].u[0] = *(const U128*)&Blds[n * BP + 8 * hh];      // K = 8h..8h+7
      bfr[tn].u[1] = *(const U128*)&Blds[n * BP + 16 + 8 * hh]; // K = 16+8h..
    }
#pragma unroll
    for (int tm = 0; tm < 2; tm++)
#pragma unroll
      for (int tn = 0; tn < 4; tn++)
        acc[tm][tn] = __builtin_amdgcn_wmma_f32_16x16x32_bf16(
            false, a[tm].v, false, bfr[tn].v, (short)0, acc[tm][tn], false, false);
    __syncthreads();
  }

  // ---- epilogue: D lane(l=N, h) VGPR r -> M = r + 8h ----
  const int hh = lane >> 4, l = lane & 15;
#pragma unroll
  for (int tn = 0; tn < 4; tn++) {
    int n = wn * 64 + tn * 16 + l;
    float bv = bias[n];
#pragma unroll
    for (int tm = 0; tm < 2; tm++) {
#pragma unroll
      for (int r = 0; r < 8; r++) {
        int m = mbase + wm * 32 + tm * 16 + 8 * hh + r;
        if (m < M) {
          float v = acc[tm][tn][r] + bv;
          if (ACT) v = silu_f(v);
          size_t o = (size_t)m * 256 + n;
          if (OUTBF) outb[o] = (__bf16)v; else outf[o] = v;
        }
      }
    }
  }
}

// ---------------------------------------------------------------------------
extern "C" void kernel_launch(void* const* d_in, const int* in_sizes, int n_in,
                              void* d_out, int out_size, void* d_ws, size_t ws_size,
                              hipStream_t stream) {
  (void)in_sizes; (void)n_in; (void)out_size; (void)ws_size;
  auto P = [&](int i) { return (const float*)d_in[i]; };
  const float* h_in  = P(0);
  const float* pos_in = P(1);
  const int*   ei    = (const int*)d_in[2];
  const float* ea_in = P(3);
  const float* nmask = P(4);
  const float* emask = P(5);
  auto bi = [](int b) { return 6 + b * 21; };  // block param base

  // ---- workspace carve-up ----
  char* cur = (char*)d_ws;
  auto alloc = [&](size_t bytes) {
    void* p = cur; cur += (bytes + 255) & ~(size_t)255; return p;
  };
  float*  h_f32  = (float*) alloc((size_t)NN * HIDD * 4);
  __bf16* h_b    = (__bf16*)alloc((size_t)NN * HIDD * 2);
  float*  agg    = (float*) alloc((size_t)NN * HIDD * 4);
  __bf16* agg_b  = (__bf16*)alloc((size_t)NN * HIDD * 2);
  __bf16* tmp_n  = (__bf16*)alloc((size_t)NN * HIDD * 2);
  float*  upd    = (float*) alloc((size_t)NN * HIDD * 4);
  __bf16* ea     = (__bf16*)alloc((size_t)NE * HIDD * 2);
  __bf16* eaT    = (__bf16*)alloc((size_t)NE * HIDD * 2);
  float*  radial = (float*) alloc((size_t)NE * 4);
  float*  cdiff  = (float*) alloc((size_t)NE * 3 * 4);
  float*  pos_ws = (float*) alloc((size_t)NN * 3 * 4);
  __bf16 *w_cW0[3], *w_cW1[3], *w_eW0[3][2], *w_eW1[3][2], *w_nW0[3][2], *w_nW1[3][2];
  for (int b = 0; b < 3; b++) {
    w_cW0[b] = (__bf16*)alloc(256 * 768 * 2);
    w_cW1[b] = (__bf16*)alloc(256 * 256 * 2);
    for (int g = 0; g < 2; g++) {
      w_eW0[b][g] = (__bf16*)alloc(256 * 768 * 2);
      w_eW1[b][g] = (__bf16*)alloc(256 * 256 * 2);
      w_nW0[b][g] = (__bf16*)alloc(256 * 512 * 2);
      w_nW1[b][g] = (__bf16*)alloc(256 * 256 * 2);
    }
  }

  auto conv = [&](const float* s, __bf16* d, int n) {
    k_f2bf<<<(n + 255) / 256, 256, 0, stream>>>(s, d, n);
  };

  // ---- weight converts (fp32 -> bf16, layout kept [dout][din] == [N][K]) ----
  for (int b = 0; b < 3; b++) {
    int B = bi(b);
    conv(P(B + 0), w_cW0[b], 256 * 768);
    conv(P(B + 2), w_cW1[b], 256 * 256);
    for (int g = 0; g < 2; g++) {
      int G = B + 5 + g * 8;
      conv(P(G + 0), w_eW0[b][g], 256 * 768);
      conv(P(G + 2), w_eW1[b][g], 256 * 256);
      conv(P(G + 4), w_nW0[b][g], 256 * 512);
      conv(P(G + 6), w_nW1[b][g], 256 * 256);
    }
  }

  // ---- input embeddings ----
  k_copy<<<(NN * 3 + 255) / 256, 256, 0, stream>>>(pos_in, pos_ws, NN * 3);
  k_node_embed<<<NN, 256, 0, stream>>>(h_in, P(73), P(74), h_f32, h_b);
  k_radial<<<(NE + 255) / 256, 256, 0, stream>>>(pos_ws, ei, radial, cdiff, ea);
  k_edge_embed<<<NE, 256, 0, stream>>>(radial, ea_in, P(69), P(70), ea);

  const int GE = (NE + 63) / 64, GN = (NN + 63) / 64;
  // ---- equivariant blocks ----
  for (int b = 0; b < 3; b++) {
    int B = bi(b);
    if (b > 0)  // recompute radial/coord_diff from updated pos; refresh ea[:,0]
      k_radial<<<(NE + 255) / 256, 256, 0, stream>>>(pos_ws, ei, radial, cdiff, ea);
    for (int g = 0; g < 2; g++) {
      int G = B + 5 + g * 8;
      // edge MLP: mij = silu(L2(silu(L1([h_r|h_c|ea]))))
      k_gemm<0, 1, 1><<<GE, 256, 0, stream>>>(NE, h_b, nullptr, ea, ei, ei + NE,
                                              w_eW0[b][g], P(G + 1), eaT, nullptr);
      k_gemm<2, 1, 1><<<GE, 256, 0, stream>>>(NE, eaT, nullptr, nullptr, nullptr, nullptr,
                                              w_eW1[b][g], P(G + 3), ea, nullptr);
      // aggregation
      k_zero<<<(NN * HIDD + 255) / 256, 256, 0, stream>>>(agg, NN * HIDD);
      k_seg<<<NE, 256, 0, stream>>>(ea, ei, emask, agg);
      conv(agg, agg_b, NN * HIDD);
      // node MLP + residual
      k_gemm<1, 1, 1><<<GN, 256, 0, stream>>>(NN, h_b, agg_b, nullptr, nullptr, nullptr,
                                              w_nW0[b][g], P(G + 5), tmp_n, nullptr);
      k_gemm<2, 0, 0><<<GN, 256, 0, stream>>>(NN, tmp_n, nullptr, nullptr, nullptr, nullptr,
                                              w_nW1[b][g], P(G + 7), nullptr, upd);
      k_node_update<<<(NN * HIDD + 255) / 256, 256, 0, stream>>>(upd, nmask, h_f32, h_b);
    }
    // coord MLP: phi = L3(silu(L2(silu(L1([h_r|h_c|mij])))))
    k_gemm<0, 1, 1><<<GE, 256, 0, stream>>>(NE, h_b, nullptr, ea, ei, ei + NE,
                                            w_cW0[b], P(B + 1), eaT, nullptr);
    k_gemm<2, 1, 1><<<GE, 256, 0, stream>>>(NE, eaT, nullptr, nullptr, nullptr, nullptr,
                                            w_cW1[b], P(B + 3), eaT, nullptr);  // in-place OK
    k_coord<<<(NE + 7) / 8, 256, 0, stream>>>(eaT, P(B + 4), ei, cdiff, emask, pos_ws);
  }

  // ---- outputs: [h (N,8) | pos (N,3) | edge_attr (E,3)] ----
  float* out = (float*)d_out;
  k_out_h<<<(NN * 8 + 255) / 256, 256, 0, stream>>>(h_f32, P(75), P(76), nmask, out);
  k_copy<<<(NN * 3 + 255) / 256, 256, 0, stream>>>(pos_ws, out + NN * 8, NN * 3);
  k_out_edge<<<(NE * 3 + 255) / 256, 256, 0, stream>>>(ea, P(71), P(72), emask,
                                                       out + NN * 8 + NN * 3);
}
